// STConvBlock_36000415875118
// MI455X (gfx1250) — matrix operand order repaired and checked by reading, not compiled, not run on hardware
//
#include <hip/hip_runtime.h>

#define NNODES 2048
#define NBATCH 4
#define EPSBN  1e-5f

typedef __attribute__((ext_vector_type(16))) __bf16 v16bf;
typedef __attribute__((ext_vector_type(8)))  float  v8f;
typedef __attribute__((ext_vector_type(4)))  unsigned int u32x4;

union FragB { v16bf v; u32x4 q[2]; };

__device__ __forceinline__ float bf2f(unsigned short h) {
    union { unsigned int u; float f; } x; x.u = ((unsigned int)h) << 16; return x.f;
}
__device__ __forceinline__ unsigned short f2bf(float f) {
    union { float f; unsigned int u; } x; x.f = f;
    unsigned int r = x.u + 0x7FFFu + ((x.u >> 16) & 1u);
    return (unsigned short)(r >> 16);
}

// ---------------------------------------------------------------------------
// 1x3 temporal conv as WMMA GEMM.  Input channels-last (B,N,Tin,CIN) bf16,
// output (B,N,Tout,64) bf16.  K-dim = 3*CIN padded to multiple of 32.
// Each wave: one (b,n,tile) -> 16 t-rows x 64 out-channels (4 WMMA acc).
// ---------------------------------------------------------------------------
template<int CIN, int TIN, int TOUT, bool RELU>
__global__ __launch_bounds__(256)
void conv1x3_wmma(const unsigned short* __restrict__ X,
                  const unsigned short* __restrict__ Wp,
                  const float* __restrict__ Bias,
                  unsigned short* __restrict__ Y)
{
    constexpr int KTOT = 3 * CIN;
    constexpr int KC   = (KTOT + 31) >> 5;
    const int lane = threadIdx.x & 31;
    const int wid  = blockIdx.x * 8 + (threadIdx.x >> 5);
    const int tile = wid & 3;
    const int n    = (wid >> 2) & (NNODES - 1);
    const int b    = wid >> 13;
    const int t0   = tile * 16;
    const int m    = lane & 15;           // A row / D column select
    const int hs   = lane >> 4;           // half-wave select
    const size_t baseBN = ((size_t)(b * NNODES + n)) * TIN * CIN;

    __builtin_prefetch(X + baseBN + (size_t)t0 * CIN, 0, 1);

    v8f zero = {0.f,0.f,0.f,0.f,0.f,0.f,0.f,0.f};
    v8f acc[4] = { zero, zero, zero, zero };

    auto loadA8 = [&](int k) -> u32x4 {   // 8 contiguous bf16 of the A fragment
        if (k >= KTOT) k = 0;             // padded K: B is zero there
        const int s = k / CIN;
        const int c = k % CIN;
        int row = t0 + s + m;
        if (row > TIN - 1) row = TIN - 1;
        return *reinterpret_cast<const u32x4*>(X + baseBN + (size_t)row * CIN + c);
    };

    for (int kc = 0; kc < KC; ++kc) {
        FragB A;
        const int kb = kc * 32 + (hs ? 8 : 0);
        A.q[0] = loadA8(kb);
        A.q[1] = loadA8(kb + 16);
#pragma unroll
        for (int nt = 0; nt < 4; ++nt) {
            FragB Bf;
            const u32x4* bp = reinterpret_cast<const u32x4*>(
                Wp + ((size_t)((nt * KC + kc) * 32 + lane)) * 16);
            Bf.q[0] = bp[0]; Bf.q[1] = bp[1];
            acc[nt] = __builtin_amdgcn_wmma_f32_16x16x32_bf16(
                false, A.v, false, Bf.v, (short)0, acc[nt], false, false);
        }
    }

    const size_t outBN = ((size_t)(b * NNODES + n)) * TOUT * 64;
    const int trow = t0 + hs * 8;
    if (t0 + 16 <= TOUT) {
        // full tile: unpredicated coalesced stores (wave-uniform branch)
#pragma unroll
        for (int nt = 0; nt < 4; ++nt) {
            const int o = nt * 16 + m;
            const float bia = Bias[o];
#pragma unroll
            for (int v = 0; v < 8; ++v) {
                float r = acc[nt][v] + bia;
                if (RELU) r = fmaxf(r, 0.f);
                Y[outBN + (size_t)(trow + v) * 64 + o] = f2bf(r);
            }
        }
    } else {
        // tail tile: per-row mask
#pragma unroll
        for (int nt = 0; nt < 4; ++nt) {
            const int o = nt * 16 + m;
            const float bia = Bias[o];
#pragma unroll
            for (int v = 0; v < 8; ++v) {
                const int t = trow + v;
                if (t < TOUT) {
                    float r = acc[nt][v] + bia;
                    if (RELU) r = fmaxf(r, 0.f);
                    Y[outBN + (size_t)t * 64 + o] = f2bf(r);
                }
            }
        }
    }
}

// ---------------------------------------------------------------------------
// Chebyshev einsum: out[(b,n,t),o] (+)= Tx[(b,n,t),c] * W[c,o], C=64, O=16.
// Accumulation carried through the WMMA C operand (fp32 Gout buffer).
// ---------------------------------------------------------------------------
__global__ __launch_bounds__(256)
void cheb_gemm_wmma(const unsigned short* __restrict__ Tx,
                    const unsigned short* __restrict__ Wp,
                    float* __restrict__ Gout, const int accum)
{
    const int lane = threadIdx.x & 31;
    const int wid  = blockIdx.x * 8 + (threadIdx.x >> 5);
    const int tile = wid & 3;
    const int n    = (wid >> 2) & (NNODES - 1);
    const int b    = wid >> 13;
    const int t0   = tile * 16;
    const int m    = lane & 15;
    const int hs   = lane >> 4;
    const bool full = (t0 + 16 <= 60);
    const size_t baseBN = ((size_t)(b * NNODES + n)) * 60 * 64;
    const size_t goutBN = ((size_t)(b * NNODES + n)) * 60 * 16;
    const int trow = t0 + hs * 8;

    v8f c = {0.f,0.f,0.f,0.f,0.f,0.f,0.f,0.f};
    if (accum) {
        if (full) {
#pragma unroll
            for (int v = 0; v < 8; ++v)
                c[v] = Gout[goutBN + (size_t)(trow + v) * 16 + m];
        } else {
#pragma unroll
            for (int v = 0; v < 8; ++v) {
                const int t = trow + v;
                if (t < 60) c[v] = Gout[goutBN + (size_t)t * 16 + m];
            }
        }
    }

    int row = t0 + m; if (row > 59) row = 59;
    for (int kc = 0; kc < 2; ++kc) {
        FragB A;
        const int kb = kc * 32 + (hs ? 8 : 0);
        A.q[0] = *reinterpret_cast<const u32x4*>(Tx + baseBN + (size_t)row * 64 + kb);
        A.q[1] = *reinterpret_cast<const u32x4*>(Tx + baseBN + (size_t)row * 64 + kb + 16);
        FragB Bf;
        const u32x4* bp = reinterpret_cast<const u32x4*>(Wp + ((size_t)(kc * 32 + lane)) * 16);
        Bf.q[0] = bp[0]; Bf.q[1] = bp[1];
        c = __builtin_amdgcn_wmma_f32_16x16x32_bf16(
            false, A.v, false, Bf.v, (short)0, c, false, false);
    }

    if (full) {
#pragma unroll
        for (int v = 0; v < 8; ++v)
            Gout[goutBN + (size_t)(trow + v) * 16 + m] = c[v];
    } else {
#pragma unroll
        for (int v = 0; v < 8; ++v) {
            const int t = trow + v;
            if (t < 60) Gout[goutBN + (size_t)t * 16 + m] = c[v];
        }
    }
}

// ---------------------------------------------------------------------------
// Graph propagation (CSR gather):  Out[b,n,:,:] = scale * sum_e norm[e]*Z[b,col,:,:] - Prev
// One block per (b,n); 256 threads cover the 60*64 = 3840-element slab.
// ---------------------------------------------------------------------------
__global__ __launch_bounds__(256)
void cheb_prop(const unsigned short* __restrict__ Z,
               const unsigned short* __restrict__ Prev,
               unsigned short* __restrict__ Out,
               const int* __restrict__ rowptr,
               const int* __restrict__ ecol,
               const float* __restrict__ enorm,
               const float scale)
{
    const int n   = blockIdx.x & (NNODES - 1);
    const int b   = blockIdx.x >> 11;
    const int tid = threadIdx.x;
    float acc[15];
#pragma unroll
    for (int j = 0; j < 15; ++j) acc[j] = 0.f;

    const int e0 = rowptr[n], e1 = rowptr[n + 1];
    for (int e = e0; e < e1; ++e) {
        const int col = ecol[e];
        const float w = enorm[e];
        const unsigned short* src = Z + ((size_t)(b * NNODES + col)) * 3840;
#pragma unroll
        for (int j = 0; j < 15; ++j) acc[j] += w * bf2f(src[tid + (j << 8)]);
    }
    unsigned short* dst = Out + ((size_t)(b * NNODES + n)) * 3840;
    const unsigned short* pv = Prev ? (Prev + ((size_t)(b * NNODES + n)) * 3840) : nullptr;
#pragma unroll
    for (int j = 0; j < 15; ++j) {
        float v = scale * acc[j];
        if (pv) v -= bf2f(pv[tid + (j << 8)]);
        dst[tid + (j << 8)] = f2bf(v);
    }
}

// ------------------------- graph CSR construction --------------------------
__global__ void zero_f32(float* p, const int n) {
    const int i = blockIdx.x * 256 + threadIdx.x; if (i < n) p[i] = 0.f;
}
__global__ void edge_deg(const int* __restrict__ ei, const float* __restrict__ ew,
                         float* degw, int* cnt, const int E) {
    const int e = blockIdx.x * 256 + threadIdx.x; if (e >= E) return;
    atomicAdd(&degw[ei[e]], ew[e]);
    atomicAdd(&cnt[ei[e]], 1);
}
__global__ void scan_dinv(const float* __restrict__ degw, const int* __restrict__ cnt,
                          int* rowptr, float* dinv) {
    if (threadIdx.x == 0 && blockIdx.x == 0) {
        int acc = 0;
        for (int n = 0; n < NNODES; ++n) {
            rowptr[n] = acc; acc += cnt[n];
            const float d = degw[n];
            dinv[n] = (d > 0.f) ? rsqrtf(d) : 0.f;
        }
        rowptr[NNODES] = acc;
    }
}
__global__ void edge_fill(const int* __restrict__ ei, const float* __restrict__ ew,
                          const int* __restrict__ rowptr, int* fcnt,
                          const float* __restrict__ dinv,
                          int* ecol, float* enorm, const int E) {
    const int e = blockIdx.x * 256 + threadIdx.x; if (e >= E) return;
    const int r = ei[e], c = ei[E + e];
    const int pos = rowptr[r] + atomicAdd(&fcnt[r], 1);
    ecol[pos] = c;
    enorm[pos] = -dinv[r] * ew[e] * dinv[c];
}

// ------------------------- BN statistics & apply ---------------------------
__global__ __launch_bounds__(256)
void stats_cl(const unsigned short* __restrict__ X, float* __restrict__ St, const long rows) {
    __shared__ float s0[256], s1[256];
    const int tid = threadIdx.x;
    const int c = tid & 63, rg = tid >> 6;
    float a = 0.f, bq = 0.f;
    for (long r = (long)blockIdx.x * 4 + rg; r < rows; r += (long)gridDim.x * 4) {
        const float v = bf2f(X[r * 64 + c]); a += v; bq += v * v;
    }
    s0[tid] = a; s1[tid] = bq; __syncthreads();
    if (tid < 64) {
        a  = s0[tid] + s0[tid + 64] + s0[tid + 128] + s0[tid + 192];
        bq = s1[tid] + s1[tid + 64] + s1[tid + 128] + s1[tid + 192];
        atomicAdd(&St[tid], a); atomicAdd(&St[64 + tid], bq);
    }
}
__global__ __launch_bounds__(256)
void stats_nchw(const float* __restrict__ X, float* __restrict__ St, const int plane) {
    __shared__ float s0[256], s1[256];
    const int bc = blockIdx.x, c = bc & 63, tid = threadIdx.x;
    const float* p = X + (size_t)bc * plane;
    float a = 0.f, bq = 0.f;
    for (int i = tid; i < plane; i += 256) { const float v = p[i]; a += v; bq += v * v; }
    s0[tid] = a; s1[tid] = bq; __syncthreads();
    for (int s = 128; s > 0; s >>= 1) {
        if (tid < s) { s0[tid] += s0[tid + s]; s1[tid] += s1[tid + s]; }
        __syncthreads();
    }
    if (tid == 0) { atomicAdd(&St[c], s0[0]); atomicAdd(&St[64 + c], s1[0]); }
}
__global__ __launch_bounds__(256)
void bn_relu_cl(const unsigned short* __restrict__ X, const float* __restrict__ St,
                const float* __restrict__ G, const float* __restrict__ Bt,
                unsigned short* __restrict__ Y, const float invCnt) {
    const long i = (long)blockIdx.x * 256 + threadIdx.x;
    const int c = (int)(i & 63);
    const float mu  = St[c] * invCnt;
    const float var = St[64 + c] * invCnt - mu * mu;
    const float inv = rsqrtf(var + EPSBN);
    const float v = (bf2f(X[i]) - mu) * inv * G[c] + Bt[c];
    Y[i] = f2bf(fmaxf(v, 0.f));
}

// ------------------------- elementwise glue --------------------------------
__global__ __launch_bounds__(256)
void pack_x(const float* __restrict__ X, unsigned short* __restrict__ Y) {
    const long i = (long)blockIdx.x * 256 + threadIdx.x;     // (b,n,t,c)
    const int c = (int)(i & 63);
    long r = i >> 6;
    const int t = (int)(r & 63); r >>= 6;
    const int n = (int)(r & (NNODES - 1));
    const int b = (int)(r >> 11);
    Y[i] = f2bf(X[(((size_t)(b * 64 + c)) * NNODES + n) * 64 + t]);
}
__global__ void pack_conv_w(const float* __restrict__ W, unsigned short* __restrict__ Wp,
                            const int CIN, const int KC) {
    const int i = blockIdx.x * 256 + threadIdx.x;
    if (i >= 4 * KC * 512) return;
    const int e = i & 15, lane = (i >> 4) & 31;
    int rest = i >> 9;
    const int kc = rest % KC, nt = rest / KC;
    const int k = kc * 32 + (lane >> 4) * 8 + (e & 7) + (e >> 3) * 16;
    float v = 0.f;
    if (k < 3 * CIN) {
        const int s = k / CIN, c = k % CIN, o = nt * 16 + (lane & 15);
        v = W[(o * CIN + c) * 3 + s];
    }
    Wp[i] = f2bf(v);
}
__global__ void pack_cheb_w(const float* __restrict__ W, unsigned short* __restrict__ Wp) {
    const int i = blockIdx.x * 256 + threadIdx.x;            // 1024 total
    if (i >= 1024) return;
    const int e = i & 15, lane = (i >> 4) & 31, kc = i >> 9;
    const int c = kc * 32 + (lane >> 4) * 8 + (e & 7) + (e >> 3) * 16;
    Wp[i] = f2bf(W[c * 16 + (lane & 15)]);
}
__global__ __launch_bounds__(256)
void gbias_relu(const float* __restrict__ Gout, const float* __restrict__ ChB,
                unsigned short* __restrict__ Gbf) {
    const long i = (long)blockIdx.x * 256 + threadIdx.x;
    const int o = (int)(i & 15);
    Gbf[i] = f2bf(fmaxf(Gout[i] + ChB[o], 0.f));
}
__global__ __launch_bounds__(256)
void bn2_resid(const unsigned short* __restrict__ H, const float* __restrict__ St,
               const float* __restrict__ G, const float* __restrict__ Bt,
               const float* __restrict__ Xin, float* __restrict__ Out,
               const float invCnt) {
    const long i = (long)blockIdx.x * 256 + threadIdx.x;     // (b,n,t,o) channels-last
    const int o = (int)(i & 63);
    long r = i >> 6;
    const int t = (int)(r % 56);
    const long bn = r / 56;
    const int n = (int)(bn & (NNODES - 1));
    const int b = (int)(bn >> 11);
    const float mu  = St[o] * invCnt;
    const float var = St[64 + o] * invCnt - mu * mu;
    const float inv = rsqrtf(var + EPSBN);
    const float h = fmaxf((bf2f(H[i]) - mu) * inv * G[o] + Bt[o], 0.f);
    const size_t base = ((size_t)(b * 64 + o)) * NNODES + n;
    Out[base * 56 + t] = h + Xin[base * 64 + t];
}
__global__ __launch_bounds__(256)
void final_bn(float* __restrict__ Out, const float* __restrict__ St,
              const float* __restrict__ G, const float* __restrict__ Bt,
              const float invCnt) {
    const long i = (long)blockIdx.x * 256 + threadIdx.x;
    const int c = (int)((i / (NNODES * 56L)) & 63);
    const float mu  = St[c] * invCnt;
    const float var = St[64 + c] * invCnt - mu * mu;
    const float inv = rsqrtf(var + EPSBN);
    Out[i] = fmaxf((Out[i] - mu) * inv * G[c] + Bt[c], 0.f);
}

// ---------------------------------------------------------------------------
extern "C" void kernel_launch(void* const* d_in, const int* in_sizes, int n_in,
                              void* d_out, int out_size, void* d_ws, size_t ws_size,
                              hipStream_t stream)
{
    const float* x    = (const float*)d_in[0];
    const int*   ei   = (const int*)  d_in[1];
    const float* ew   = (const float*)d_in[2];
    const float* t1w1 = (const float*)d_in[3];  const float* t1b1 = (const float*)d_in[4];
    const float* t1w2 = (const float*)d_in[5];  const float* t1b2 = (const float*)d_in[6];
    const float* bn1g = (const float*)d_in[7];  const float* bn1b = (const float*)d_in[8];
    const float* chW  = (const float*)d_in[9];  const float* chB  = (const float*)d_in[10];
    const float* t2w1 = (const float*)d_in[11]; const float* t2b1 = (const float*)d_in[12];
    const float* t2w2 = (const float*)d_in[13]; const float* t2b2 = (const float*)d_in[14];
    const float* bn2g = (const float*)d_in[15]; const float* bn2b = (const float*)d_in[16];
    const float* bnfg = (const float*)d_in[17]; const float* bnfb = (const float*)d_in[18];
    float* out = (float*)d_out;
    const int E = in_sizes[2];

    char* ws = (char*)d_ws;
    size_t off = 0;
    auto alloc = [&](size_t bytes) -> char* {
        char* p = ws + off; off += (bytes + 255) & ~(size_t)255; return p;
    };
    // Region reuse plan:  R1: xcl -> c2raw -> Tx2 -> g   R2: h1 -> Tx1 -> h3   R3: Tx0 -> h4
    unsigned short* R1   = (unsigned short*)alloc(67108864);
    unsigned short* R2   = (unsigned short*)alloc(65011712);
    unsigned short* R3   = (unsigned short*)alloc(62914560);
    float*          gout = (float*)         alloc(31457280);
    unsigned short* wp1  = (unsigned short*)alloc(24576);
    unsigned short* wp2  = (unsigned short*)alloc(24576);
    unsigned short* wp3  = (unsigned short*)alloc(8192);
    unsigned short* wp4  = (unsigned short*)alloc(24576);
    unsigned short* wpc  = (unsigned short*)alloc(3 * 2048);
    float* st    = (float*)alloc(3 * 128 * sizeof(float));   // st1|st2|st3
    float* degw  = (float*)alloc(NNODES * 4);
    float* dinv  = (float*)alloc(NNODES * 4);
    int*   cnt   = (int*)  alloc(NNODES * 4);
    int*   fcnt  = (int*)  alloc(NNODES * 4);
    int*   rowp  = (int*)  alloc((NNODES + 1) * 4);
    int*   ecol  = (int*)  alloc((size_t)E * 4);
    float* enorm = (float*)alloc((size_t)E * 4);

    float* st1 = st; float* st2 = st + 128; float* st3 = st + 256;
    const float ic1 = 1.f / (4.f * NNODES * 60.f);
    const float ic2 = 1.f / (4.f * NNODES * 56.f);

    // --- init / packing ---
    zero_f32<<<2, 256, 0, stream>>>(st, 384);
    zero_f32<<<8, 256, 0, stream>>>(degw, NNODES);
    zero_f32<<<8, 256, 0, stream>>>((float*)cnt, NNODES);
    zero_f32<<<8, 256, 0, stream>>>((float*)fcnt, NNODES);
    pack_x<<<131072, 256, 0, stream>>>(x, R1);                      // xcl
    pack_conv_w<<<48, 256, 0, stream>>>(t1w1, wp1, 64, 6);
    pack_conv_w<<<48, 256, 0, stream>>>(t1w2, wp2, 64, 6);
    pack_conv_w<<<16, 256, 0, stream>>>(t2w1, wp3, 16, 2);
    pack_conv_w<<<48, 256, 0, stream>>>(t2w2, wp4, 64, 6);
    for (int k = 0; k < 3; ++k)
        pack_cheb_w<<<4, 256, 0, stream>>>(chW + k * 64 * 16, wpc + k * 1024);

    // --- temporal block 1 ---
    conv1x3_wmma<64, 64, 62, true ><<<4096, 256, 0, stream>>>(R1, wp1, t1b1, R2); // h1
    conv1x3_wmma<64, 62, 60, false><<<4096, 256, 0, stream>>>(R2, wp2, t1b2, R1); // c2raw
    stats_cl<<<2048, 256, 0, stream>>>(R1, st1, 4L * NNODES * 60);
    bn_relu_cl<<<122880, 256, 0, stream>>>(R1, st1, bn1g, bn1b, R3, ic1);         // Tx0

    // --- graph CSR ---
    edge_deg <<<(E + 255) / 256, 256, 0, stream>>>(ei, ew, degw, cnt, E);
    scan_dinv<<<1, 1, 0, stream>>>(degw, cnt, rowp, dinv);
    edge_fill<<<(E + 255) / 256, 256, 0, stream>>>(ei, ew, rowp, fcnt, dinv, ecol, enorm, E);

    // --- Chebyshev recurrence + einsum accumulation ---
    cheb_gemm_wmma<<<4096, 256, 0, stream>>>(R3, wpc,          gout, 0);          // k=0
    cheb_prop     <<<8192, 256, 0, stream>>>(R3, nullptr, R2, rowp, ecol, enorm, 1.f); // Tx1
    cheb_gemm_wmma<<<4096, 256, 0, stream>>>(R2, wpc + 1024,   gout, 1);          // k=1
    cheb_prop     <<<8192, 256, 0, stream>>>(R2, R3,      R1, rowp, ecol, enorm, 2.f); // Tx2
    cheb_gemm_wmma<<<4096, 256, 0, stream>>>(R1, wpc + 2048,   gout, 1);          // k=2
    gbias_relu    <<<30720, 256, 0, stream>>>(gout, chB, R1);                     // g bf16

    // --- temporal block 2 ---
    conv1x3_wmma<16, 60, 58, true ><<<4096, 256, 0, stream>>>(R1, wp3, t2b1, R2); // h3
    conv1x3_wmma<64, 58, 56, false><<<4096, 256, 0, stream>>>(R2, wp4, t2b2, R3); // h4
    stats_cl<<<2048, 256, 0, stream>>>(R3, st2, 4L * NNODES * 56);
    bn2_resid<<<114688, 256, 0, stream>>>(R3, st2, bn2g, bn2b, x, out, ic2);

    // --- final BN + ReLU (in place on d_out, NCHW) ---
    stats_nchw<<<256, 256, 0, stream>>>(out, st3, NNODES * 56);
    final_bn<<<114688, 256, 0, stream>>>(out, st3, bnfg, bnfb, ic2);

    (void)n_in; (void)out_size; (void)ws_size;
}
